// InfoNCELoss_46746424049928
// MI455X (gfx1250) — matrix operand order repaired and checked by reading, not compiled, not run on hardware
//
#include <hip/hip_runtime.h>

// ---------------------------------------------------------------------------
// InfoNCE loss, fused flash-style for MI455X (gfx1250, wave32, WMMA).
//
// sim = (E/||E||)(E/||E||)^T / 0.07 (8192x8192) is never materialized.
// GEMM via V_WMMA_F32_16X16X32_BF16 with a 3-term bf16 split of the
// normalized fp32 embeddings (hi*hi + hi*lo + lo*hi): ~fp32 accuracy at
// bf16-WMMA rate. Normalized hi+lo matrices total 16 MB -> L2-resident
// (192 MB L2), so B-operand re-reads hit L2; the 64 MB bool mask streams
// once with a non-temporal hint so it doesn't evict the embeddings.
//
// LSE uses a FIXED max M = 1/T: cosine similarity / T is bounded by 1/T,
// so exp(x - M) ∈ [e^-28.6, 1] — no fp32 under/overflow. This removes the
// running-max chain and halves the transcendental ops in the inner tail
// (exps share the TRANS pipe with 16-bit WMMA on CDNA5).
// ---------------------------------------------------------------------------

typedef __bf16 bf16_t;
typedef __attribute__((ext_vector_type(16))) __bf16 v16bf;
typedef __attribute__((ext_vector_type(4)))  __bf16 v4bf;
typedef __attribute__((ext_vector_type(8)))  float  v8f;

#define B_N 8192
#define D_K 512
#define INV_T (1.0f / 0.07f)
#define MASKV (-10000.0f)
#define LSE_MAX INV_T   // static upper bound on sim/T (cos sim <= 1)

union FragB16 {
    v16bf v;
    uint4 q[2];
};

// ---------------------------------------------------------------------------
// Kernel 1: L2-normalize each row (eps clamp like F.normalize) and split each
// fp32 value into bf16 hi + bf16 lo (lo = RNE(x - hi)). One wave per row.
// ---------------------------------------------------------------------------
__global__ __launch_bounds__(256) void norm_split_kernel(
    const float* __restrict__ E, bf16_t* __restrict__ Ehi, bf16_t* __restrict__ Elo)
{
    const int lane = threadIdx.x & 31;
    const int wid  = threadIdx.x >> 5;
    const int row  = blockIdx.x * 8 + wid;

    const float* r = E + (size_t)row * D_K;

    float4 vals[4];
    float ss = 0.0f;
#pragma unroll
    for (int c = 0; c < 4; ++c) {
        vals[c] = *(const float4*)(r + c * 128 + lane * 4);
        ss += vals[c].x * vals[c].x + vals[c].y * vals[c].y +
              vals[c].z * vals[c].z + vals[c].w * vals[c].w;
    }
#pragma unroll
    for (int off = 16; off > 0; off >>= 1)
        ss += __shfl_xor(ss, off, 32);

    const float scale = 1.0f / fmaxf(sqrtf(ss), 1e-12f);

#pragma unroll
    for (int c = 0; c < 4; ++c) {
        float f[4] = { vals[c].x * scale, vals[c].y * scale,
                       vals[c].z * scale, vals[c].w * scale };
        v4bf h, l;
#pragma unroll
        for (int i = 0; i < 4; ++i) {
            bf16_t hv = (bf16_t)f[i];
            h[i] = hv;
            l[i] = (bf16_t)(f[i] - (float)hv);
        }
        const size_t idx = (size_t)row * D_K + c * 128 + lane * 4;
        *(v4bf*)(Ehi + idx) = h;
        *(v4bf*)(Elo + idx) = l;
    }
}

// ---------------------------------------------------------------------------
// Kernel 2: fused sim-tile GEMM + diag mask + positive sum/count + fixed-max
// LSE accumulation. Block = 256 threads = 8 waves as 2 (row-tiles) x 4
// (col-tiles): 32 rows x full 8192-column sweep (128 iters of 32x64 tiles).
// Per 16x16 tile: K=512 via 16 k-steps x 3 bf16 WMMAs.
// ---------------------------------------------------------------------------
__global__ __launch_bounds__(256) void infonce_main_kernel(
    const bf16_t* __restrict__ Ehi, const bf16_t* __restrict__ Elo,
    const unsigned char* __restrict__ mask, float* __restrict__ rowloss)
{
    __shared__ float red[3][8][8][32];   // [S|PS|PC][wave][acc-slot][lane] : 24 KB

    const int tid  = threadIdx.x;
    const int lane = tid & 31;
    const int wid  = tid >> 5;
    const int wrow = wid >> 2;         // 0..1 : which 16-row tile
    const int wcol = wid & 3;          // 0..3 : which 16-col tile in a 64-col group
    const int r0   = blockIdx.x * 32 + wrow * 16;
    const int lm   = lane & 15;
    const int lh   = lane >> 4;

    // ---- Preload A fragments: rows r0..r0+15, hi & lo, all 16 k-steps -----
    // A 16x32 bf16 layout: lane holds row (lane&15);
    // VGPR0-7 = K[kh..kh+15], kh = (lane>>4)*16 (matches B's contiguous order).
    FragB16 Ahi[16], Alo[16];
    {
        const bf16_t* rh = Ehi + (size_t)(r0 + lm) * D_K;
        const bf16_t* rl = Elo + (size_t)(r0 + lm) * D_K;
        const int kh = lh * 16;
#pragma unroll
        for (int ks = 0; ks < 16; ++ks) {
            const int k = ks * 32;
            Ahi[ks].q[0] = *(const uint4*)(rh + k + kh);
            Ahi[ks].q[1] = *(const uint4*)(rh + k + kh + 8);
            Alo[ks].q[0] = *(const uint4*)(rl + k + kh);
            Alo[ks].q[1] = *(const uint4*)(rl + k + kh + 8);
        }
    }

    // Per-slot accumulators: sum of exp(x - LSE_MAX), positive sum, pos count.
    float S[8], PS[8], PC[8];
#pragma unroll
    for (int v = 0; v < 8; ++v) { S[v] = 0.f; PS[v] = 0.f; PC[v] = 0.f; }

    const int boff = lh * 16;  // B 32x16 layout: lane holds col (lane&15), K = boff..boff+15

    for (int it = 0; it < B_N / 64; ++it) {
        const int j0 = it * 64 + wcol * 16;
        const bf16_t* bh = Ehi + (size_t)(j0 + lm) * D_K + boff;
        const bf16_t* bl = Elo + (size_t)(j0 + lm) * D_K + boff;

        v8f acc = {};
#pragma unroll
        for (int ks = 0; ks < 16; ++ks) {
            const int k = ks * 32;
            FragB16 Bh, Bl;
            Bh.q[0] = *(const uint4*)(bh + k);
            Bh.q[1] = *(const uint4*)(bh + k + 8);
            Bl.q[0] = *(const uint4*)(bl + k);
            Bl.q[1] = *(const uint4*)(bl + k + 8);
            acc = __builtin_amdgcn_wmma_f32_16x16x32_bf16(
                false, Ahi[ks].v, false, Bh.v, (short)0, acc, false, false);
            acc = __builtin_amdgcn_wmma_f32_16x16x32_bf16(
                false, Ahi[ks].v, false, Bl.v, (short)0, acc, false, false);
            acc = __builtin_amdgcn_wmma_f32_16x16x32_bf16(
                false, Alo[ks].v, false, Bh.v, (short)0, acc, false, false);
        }

        // C/D layout: VGPR v, lanes 0-15 -> (M=v, N=lane); lanes 16-31 -> (M=v+8).
        const int col = j0 + lm;
#pragma unroll
        for (int v = 0; v < 8; ++v) {
            const int row = r0 + v + lh * 8;
            float x = acc[v] * INV_T;
            const bool diag = (row == col);
            x = diag ? MASKV : x;
            // Non-temporal: mask streams exactly once, keep L2 for embeddings.
            const unsigned char mb =
                __builtin_nontemporal_load(mask + (size_t)row * B_N + col);
            const float p = (!diag && mb != 0) ? 1.0f : 0.0f;
            PS[v] += p * x;
            PC[v] += p;
            S[v]  += __expf(x - LSE_MAX);  // exp(-10000-M) flushes to exactly 0
        }
    }

    // ---- Cross-lane / cross-wave merge via LDS (plain sums) ---------------
#pragma unroll
    for (int v = 0; v < 8; ++v) {
        red[0][wid][v][lane] = S[v];
        red[1][wid][v][lane] = PS[v];
        red[2][wid][v][lane] = PC[v];
    }
    __syncthreads();

    if (tid < 32) {
        const int r    = tid;             // row within block (0..31)
        const int wr   = r >> 4;          // which row-tile wave group
        const int m    = r & 15;
        const int v    = m & 7;
        const int half = (m >> 3) * 16;   // lanes holding this row
        float Ss = 0.f, ps = 0.f, pc = 0.f;
        for (int wc = 0; wc < 4; ++wc) {
            const int w = wr * 4 + wc;
#pragma unroll
            for (int l = 0; l < 16; ++l) {
                Ss += red[0][w][v][half + l];
                ps += red[1][w][v][half + l];
                pc += red[2][w][v][half + l];
            }
        }
        const float lse  = LSE_MAX + __logf(Ss);
        const float loss = (pc > 0.f) ? (lse - ps / pc) : lse;
        rowloss[blockIdx.x * 32 + r] = loss;
    }
}

// ---------------------------------------------------------------------------
// Kernel 3: deterministic mean of the 8192 per-row losses.
// ---------------------------------------------------------------------------
__global__ __launch_bounds__(256) void reduce_mean_kernel(
    const float* __restrict__ rowloss, float* __restrict__ out)
{
    __shared__ float sm[256];
    float s = 0.0f;
    for (int i = threadIdx.x; i < B_N; i += 256)
        s += rowloss[i];
    sm[threadIdx.x] = s;
    __syncthreads();
    for (int st = 128; st > 0; st >>= 1) {
        if (threadIdx.x < st) sm[threadIdx.x] += sm[threadIdx.x + st];
        __syncthreads();
    }
    if (threadIdx.x == 0)
        out[0] = sm[0] / (float)B_N;
}

// ---------------------------------------------------------------------------
extern "C" void kernel_launch(void* const* d_in, const int* in_sizes, int n_in,
                              void* d_out, int out_size, void* d_ws, size_t ws_size,
                              hipStream_t stream)
{
    const float*         E    = (const float*)d_in[0];
    const unsigned char* mask = (const unsigned char*)d_in[1];  // jnp bool = 1 byte

    char* ws = (char*)d_ws;
    bf16_t* Ehi     = (bf16_t*)ws;                                   //  8 MB
    bf16_t* Elo     = (bf16_t*)(ws + (size_t)B_N * D_K * 2);         //  8 MB
    float*  rowloss = (float*) (ws + (size_t)B_N * D_K * 4);         // 32 KB

    norm_split_kernel  <<<B_N / 8,  256, 0, stream>>>(E, Ehi, Elo);
    infonce_main_kernel<<<B_N / 32, 256, 0, stream>>>(Ehi, Elo, mask, rowloss);
    reduce_mean_kernel <<<1,        256, 0, stream>>>(rowloss, (float*)d_out);
}
